// GNN_5592047419732
// MI455X (gfx1250) — matrix operand order repaired and checked by reading, not compiled, not run on hardware
//
#include <hip/hip_runtime.h>
#include <hip/hip_bf16.h>
#include <math.h>

#define N_NODES 50000
#define N_EDGES 800000
#define DIM     256
#define DEPTH   4
#define N_REL   4
#define M_WAVES 3125          // 50000 / 16 wave-tiles along M

typedef __attribute__((ext_vector_type(16))) __bf16 v16bf;
typedef __attribute__((ext_vector_type(8)))  float  v8f;

// ---------------------------------------------------------------------------
// helpers
// ---------------------------------------------------------------------------
static __device__ __forceinline__ unsigned short f2bf_bits(float v) {
  unsigned u = __float_as_uint(v);
  return (unsigned short)((u + 0x7FFFu + ((u >> 16) & 1u)) >> 16);  // RNE
}

// ---------------------------------------------------------------------------
// 1) per-(node,relation) in-degree count -> inv_deg per edge
// ---------------------------------------------------------------------------
__global__ __launch_bounds__(256) void count_kernel(const int* __restrict__ dst,
                                                    const int* __restrict__ et,
                                                    int* __restrict__ cnt) {
  int e = blockIdx.x * 256 + threadIdx.x;
  if (e < N_EDGES) atomicAdd(&cnt[dst[e] * N_REL + et[e]], 1);
}

__global__ __launch_bounds__(256) void invdeg_kernel(const int* __restrict__ dst,
                                                     const int* __restrict__ et,
                                                     const int* __restrict__ cnt,
                                                     float* __restrict__ invd) {
  int e = blockIdx.x * 256 + threadIdx.x;
  if (e < N_EDGES) {
    float c = (float)cnt[dst[e] * N_REL + et[e]];
    invd[e] = 1.0f / fmaxf(c, 1.0f);
  }
}

// ---------------------------------------------------------------------------
// 2) pack weights [DEPTH][5][DIM(k)][DIM(n)] f32 -> bf16 WMMA B-fragments
//    layout: [l][g][k8][ntile][lane][16 elems], 32B contiguous per lane
//    lane holds col n = ntile*16+(lane&15), K = k8*32 + 16*(lane>>4) + e
// ---------------------------------------------------------------------------
__global__ __launch_bounds__(256) void pack_kernel(const float* __restrict__ Wts,
                                                   const float* __restrict__ Rts,
                                                   unsigned short* __restrict__ pW) {
  int tid  = blockIdx.x * 256 + threadIdx.x;           // enumerates (l,g,k8,ntile,lane)
  int lane = tid & 31;
  int t    = tid >> 5;
  int ntile = t & 15;  t >>= 4;
  int k8    = t & 7;   t >>= 3;
  int g     = t % 5;
  int l     = t / 5;
  int n     = ntile * 16 + (lane & 15);
  int kbase = k8 * 32 + ((lane >> 4) << 4);
  unsigned short* d = pW + (size_t)tid * 16;           // mixed radix matches address
#pragma unroll
  for (int e = 0; e < 16; ++e) {
    int k = kbase + e;
    float v = (g < 4) ? Wts[((size_t)(l * N_REL + g) * DIM + k) * DIM + n]
                      : Rts[((size_t)l * DIM + k) * DIM + n];
    d[e] = f2bf_bits(v);
  }
}

// ---------------------------------------------------------------------------
// 3) edge scatter: A[r][dst] += X[src] * inv_deg   (f32 atomics, L2-resident X)
//    4 edges / block, 64 threads per edge, float4 per thread
// ---------------------------------------------------------------------------
__global__ __launch_bounds__(256) void agg_kernel(const float* __restrict__ X,
                                                  const int* __restrict__ src,
                                                  const int* __restrict__ dst,
                                                  const int* __restrict__ et,
                                                  const float* __restrict__ invd,
                                                  float* __restrict__ A) {
  int e   = blockIdx.x * 4 + (threadIdx.x >> 6);
  int sub = threadIdx.x & 63;
  int s = src[e], d = dst[e], r = et[e];
  float w = invd[e];
  float4 v = ((const float4*)(X + (size_t)s * DIM))[sub];
  float* ap = A + ((size_t)r * N_NODES + d) * DIM + sub * 4;
  __hip_atomic_fetch_add(ap + 0, v.x * w, __ATOMIC_RELAXED, __HIP_MEMORY_SCOPE_AGENT);
  __hip_atomic_fetch_add(ap + 1, v.y * w, __ATOMIC_RELAXED, __HIP_MEMORY_SCOPE_AGENT);
  __hip_atomic_fetch_add(ap + 2, v.z * w, __ATOMIC_RELAXED, __HIP_MEMORY_SCOPE_AGENT);
  __hip_atomic_fetch_add(ap + 3, v.w * w, __ATOMIC_RELAXED, __HIP_MEMORY_SCOPE_AGENT);
}

// ---------------------------------------------------------------------------
// 4) fused GEMM + bias + LayerNorm + ELU + residual
//    Wave-per-M-stripe tiling: wave-tile wid = blockIdx*8 + waveId owns rows
//    [wid*16, wid*16+16) x all 256 cols (16 WMMA n-tiles, acc = 16 x v8f).
//    A is read exactly once; B (655 KB/layer) is L2-resident and shared.
//    LayerNorm is fully in-wave (shfl_xor over 16-lane halves): no LDS, no
//    barriers, so tail waves simply exit (EXEC stays all-ones for WMMA).
// ---------------------------------------------------------------------------
__global__ __launch_bounds__(256) void rgcn_gemm_kernel(const float* __restrict__ A,
                                                        const float* __restrict__ Xin,
                                                        const unsigned short* __restrict__ pW,
                                                        const float* __restrict__ bias,
                                                        const float* __restrict__ gamma,
                                                        const float* __restrict__ beta,
                                                        float* __restrict__ Xout) {
  const int lane = threadIdx.x & 31;
  const int wid  = blockIdx.x * 8 + (threadIdx.x >> 5);   // wave-tile id along M
  if (wid >= M_WAVES) return;                             // wave-uniform tail exit

  const int l15 = lane & 15;
  const int kb  = (lane >> 4) << 4;                       // A-frag K split: 0 or 16
  const size_t rowA = (size_t)(wid * 16 + l15) * DIM;     // this lane's A row

  v8f acc[16];
#pragma unroll
  for (int nt = 0; nt < 16; ++nt) acc[nt] = (v8f){0.f,0.f,0.f,0.f,0.f,0.f,0.f,0.f};

  const v16bf* pWv = (const v16bf*)pW;

  for (int s = 0; s < 40; ++s) {
    const int g  = s >> 3;                 // 0..3 = A_r, 4 = X
    const int kk = (s & 7) << 5;           // K offset within the 256-wide matrix
    const float* srcX = (g < 4) ? (A + (size_t)g * N_NODES * DIM) : Xin;
    const float* pr   = srcX + rowA + kk + kb;

    if (s < 39) __builtin_prefetch(pr + 32, 0, 0);        // next k-step's A data

    float4 f0 = ((const float4*)pr)[0];
    float4 f1 = ((const float4*)pr)[1];
    float4 f2 = ((const float4*)(pr + 8))[0];
    float4 f3 = ((const float4*)(pr + 8))[1];

    v16bf a;
    a[0]  = (__bf16)f0.x; a[1]  = (__bf16)f0.y; a[2]  = (__bf16)f0.z; a[3]  = (__bf16)f0.w;
    a[4]  = (__bf16)f1.x; a[5]  = (__bf16)f1.y; a[6]  = (__bf16)f1.z; a[7]  = (__bf16)f1.w;
    a[8]  = (__bf16)f2.x; a[9]  = (__bf16)f2.y; a[10] = (__bf16)f2.z; a[11] = (__bf16)f2.w;
    a[12] = (__bf16)f3.x; a[13] = (__bf16)f3.y; a[14] = (__bf16)f3.z; a[15] = (__bf16)f3.w;

    const int base = (g * 8 + (s & 7)) * 16;
#pragma unroll
    for (int nt = 0; nt < 16; ++nt) {
      v16bf b = pWv[(size_t)(base + nt) * 32 + lane];
      acc[nt] = __builtin_amdgcn_wmma_f32_16x16x32_bf16(false, a, false, b, (short)0,
                                                        acc[nt], false, false);
    }
  }

  // ---- epilogue: bias + LayerNorm + ELU + residual, fully in-wave ----
  // C layout: lane element j -> row = j + 8*(lane>>4), col = nt*16 + (lane&15)
#pragma unroll
  for (int nt = 0; nt < 16; ++nt) {
    float b = bias[nt * 16 + l15];
#pragma unroll
    for (int j = 0; j < 8; ++j) acc[nt][j] += b;
  }

  float sm[8], sq[8];
#pragma unroll
  for (int j = 0; j < 8; ++j) { sm[j] = 0.f; sq[j] = 0.f; }
#pragma unroll
  for (int nt = 0; nt < 16; ++nt)
#pragma unroll
    for (int j = 0; j < 8; ++j) { float v = acc[nt][j]; sm[j] += v; sq[j] += v * v; }

  // reduce across the 16 lanes of each half (xor masks < 16 stay in the half)
#pragma unroll
  for (int mask = 1; mask < 16; mask <<= 1) {
#pragma unroll
    for (int j = 0; j < 8; ++j) {
      sm[j] += __shfl_xor(sm[j], mask, 32);
      sq[j] += __shfl_xor(sq[j], mask, 32);
    }
  }

  float mu[8], rs[8];
#pragma unroll
  for (int j = 0; j < 8; ++j) {
    mu[j] = sm[j] * (1.0f / DIM);
    float var = sq[j] * (1.0f / DIM) - mu[j] * mu[j];
    rs[j] = rsqrtf(var + 1e-5f);
  }

  const int rbase = wid * 16 + ((lane >> 4) << 3);
#pragma unroll
  for (int nt = 0; nt < 16; ++nt) {
    int col = nt * 16 + l15;
    float gm = gamma[col], bt = beta[col];
#pragma unroll
    for (int j = 0; j < 8; ++j) {
      size_t idx = (size_t)(rbase + j) * DIM + col;
      float v = (acc[nt][j] - mu[j]) * rs[j] * gm + bt;
      v = v > 0.f ? v : (__expf(v) - 1.0f);             // ELU
      Xout[idx] = v + Xin[idx];                         // residual
    }
  }
}

// ---------------------------------------------------------------------------
// launcher
// ---------------------------------------------------------------------------
extern "C" void kernel_launch(void* const* d_in, const int* in_sizes, int n_in,
                              void* d_out, int out_size, void* d_ws, size_t ws_size,
                              hipStream_t stream) {
  (void)in_sizes; (void)n_in; (void)out_size; (void)ws_size;

  const float* X0   = (const float*)d_in[0];   // [N, D]
  const float* Wts  = (const float*)d_in[1];   // [DEPTH, R, D, D]
  const float* Rts  = (const float*)d_in[2];   // [DEPTH, D, D]
  const float* Bias = (const float*)d_in[3];   // [DEPTH, D]
  const float* Gam  = (const float*)d_in[4];   // [DEPTH, D]
  const float* Bet  = (const float*)d_in[5];   // [DEPTH, D]
  const int*   EI   = (const int*)d_in[6];     // [2, E]
  const int*   ET   = (const int*)d_in[7];     // [E]
  float* out = (float*)d_out;

  char* ws = (char*)d_ws;
  size_t off = 0;
  float* Aagg = (float*)(ws + off);  off += (size_t)N_REL * N_NODES * DIM * 4;   // 204.8 MB
  float* Xping = (float*)(ws + off); off += (size_t)N_NODES * DIM * 4;           // 51.2 MB
  unsigned short* pW = (unsigned short*)(ws + off);
  off += (size_t)DEPTH * 5 * 8 * 16 * 32 * 16 * 2;                               // 2.6 MB
  int* cnt = (int*)(ws + off);       off += (size_t)N_NODES * N_REL * 4;         // 0.8 MB
  float* invd = (float*)(ws + off);  off += (size_t)N_EDGES * 4;                 // 3.2 MB

  const int* src = EI;
  const int* dst = EI + N_EDGES;

  // one-time prep
  hipMemsetAsync(cnt, 0, (size_t)N_NODES * N_REL * 4, stream);
  count_kernel<<<(N_EDGES + 255) / 256, 256, 0, stream>>>(dst, ET, cnt);
  invdeg_kernel<<<(N_EDGES + 255) / 256, 256, 0, stream>>>(dst, ET, cnt, invd);
  pack_kernel<<<(DEPTH * 5 * 8 * 16 * 32) / 256, 256, 0, stream>>>(Wts, Rts, pW);

  const float* Xs = X0;
  const int gemm_blocks = (M_WAVES + 7) / 8;             // 391 blocks x 8 waves
  for (int l = 0; l < DEPTH; ++l) {
    float* Xd = (l & 1) ? out : Xping;                   // L0->ws, L1->out, L2->ws, L3->out
    hipMemsetAsync(Aagg, 0, (size_t)N_REL * N_NODES * DIM * 4, stream);
    agg_kernel<<<N_EDGES / 4, 256, 0, stream>>>(Xs, src, dst, ET, invd, Aagg);
    rgcn_gemm_kernel<<<gemm_blocks, 256, 0, stream>>>(
        Aagg, Xs, pW + (size_t)l * 5 * 8 * 16 * 32 * 16,
        Bias + l * DIM, Gam + l * DIM, Bet + l * DIM, Xd);
    Xs = Xd;
  }
}
